// GATDotClassifier_45861660787102
// MI455X (gfx1250) — compile-verified, matrix operand order checked by reading, not run on hardware
//
#include <hip/hip_runtime.h>
#include <hip/hip_bf16.h>
#include <math.h>

#define NNODES 20000
#define NEDGES 320000
#define IN_DIM 128
#define HID 64
#define HEADS 4
#define NUM_GRAPHS 128
#define N_CLASSES 10

typedef __attribute__((ext_vector_type(16))) __bf16 v16bf;
typedef __attribute__((ext_vector_type(8)))  __bf16 v8bf;
typedef __attribute__((ext_vector_type(8)))  float  v8f;

// ---------------- utility ----------------
__global__ void fill_kernel(float* __restrict__ p, float v, int n) {
    int i = blockIdx.x * blockDim.x + threadIdx.x;
    if (i < n) p[i] = v;
}

// float atomic max via signed/unsigned int ordering trick (works with -inf init)
__device__ inline void atomicMaxF(float* addr, float val) {
    int iv = __float_as_int(val);
    if (iv >= 0) atomicMax((int*)addr, iv);
    else         atomicMin((unsigned int*)addr, (unsigned int)iv);
}

// ---------------- precision / layout conversion ----------------
template<bool RELU>
__global__ void f32_to_bf16(const float* __restrict__ in, __bf16* __restrict__ out,
                            long long n) {
    long long i = (long long)blockIdx.x * blockDim.x + threadIdx.x;
    if (i >= n) return;
    float v = in[i];
    if (RELU) v = fmaxf(v, 0.0f);
    out[i] = (__bf16)v;
}

// Bt[n*K + k] = bf16(B[k*Nc + n]) : transpose so every WMMA B fragment is a
// contiguous 16-byte load (same addressing pattern as the A fragment).
__global__ void transpose_to_bf16(const float* __restrict__ B, __bf16* __restrict__ Bt,
                                  int K, int Nc) {
    int i = blockIdx.x * blockDim.x + threadIdx.x;
    if (i >= K * Nc) return;
    int k = i / Nc, n = i % Nc;
    Bt[(size_t)n * K + k] = (__bf16)B[i];
}

// ---------------- WMMA GEMM: C[M,Nc] = A[M,K] @ B[K,Nc], bf16 in / f32 acc ----
// One wave32 computes a 16x64 strip (4 tiles). Per 32-wide K chunk:
//   A fragment: lanes 0-15 row M=r hold K = k0+[0..7] ++ k0+[16..23];
//               lanes 16-31 same row hold K = k0+[8..15] ++ k0+[24..31]
//   -> two contiguous v8bf (b128) loads per lane. B identical via Bt.
// D layout: VGPR v -> row = 16*tileM + v + 8*(lane>=16), col = tile base + (lane&15).
__global__ void wmma_gemm_bf16(const __bf16* __restrict__ A,   // [M,K] row-major
                               const __bf16* __restrict__ Bt,  // [Nc,K]
                               float* __restrict__ C, int M, int K, int Nc) {
    int wave   = blockIdx.x * (blockDim.x >> 5) + (threadIdx.x >> 5);
    int strips = Nc >> 6;                       // 64-wide output strips
    int tileM  = wave / strips;
    int strip  = wave % strips;
    if (tileM * 16 >= M) return;                // wave-uniform (EXEC stays all-1)

    int lane = threadIdx.x & 31;
    int half = lane >> 4;
    int r    = lane & 15;
    const __bf16* __restrict__ arow = A + (size_t)(tileM * 16 + r) * K;

    v8f acc[4] = {{}, {}, {}, {}};
    for (int k0 = 0; k0 < K; k0 += 32) {
        v16bf a;
        {
            v8bf lo = *(const v8bf*)(arow + k0 + 8 * half);
            v8bf hi = *(const v8bf*)(arow + k0 + 16 + 8 * half);
#pragma unroll
            for (int j = 0; j < 8; ++j) { a[j] = lo[j]; a[j + 8] = hi[j]; }
        }
#pragma unroll
        for (int t = 0; t < 4; ++t) {
            const __bf16* __restrict__ brow =
                Bt + (size_t)(strip * 64 + t * 16 + r) * K;
            v16bf b;
            v8bf lo = *(const v8bf*)(brow + k0 + 8 * half);
            v8bf hi = *(const v8bf*)(brow + k0 + 16 + 8 * half);
#pragma unroll
            for (int j = 0; j < 8; ++j) { b[j] = lo[j]; b[j + 8] = hi[j]; }
            acc[t] = __builtin_amdgcn_wmma_f32_16x16x32_bf16(
                false, a, false, b, (short)0, acc[t], false, false);
        }
    }
#pragma unroll
    for (int t = 0; t < 4; ++t)
#pragma unroll
        for (int vv = 0; vv < 8; ++vv) {
            int row = tileM * 16 + vv + 8 * half;
            C[(size_t)row * Nc + strip * 64 + t * 16 + r] = acc[t][vv];
        }
}

// ---------------- edge attention kernels ----------------
// e = dot(ft[src], ft[dst]) / sqrt(D) per head; atomic segment-max over dst.
// float4 loads -> b128 traffic out of L2.
__global__ void edge_scores(const float* __restrict__ ft, const int* __restrict__ src,
                            const int* __restrict__ dst, float* __restrict__ sc,
                            float* __restrict__ mmax, int E, int heads, int D) {
    int idx = blockIdx.x * blockDim.x + threadIdx.x;
    if (idx >= E * heads) return;
    int e = idx / heads, h = idx % heads;
    int s = src[e], d = dst[e];
    const float4* __restrict__ ps = (const float4*)(ft + ((size_t)s * heads + h) * D);
    const float4* __restrict__ pd = (const float4*)(ft + ((size_t)d * heads + h) * D);
    float acc = 0.0f;
#pragma unroll
    for (int j = 0; j < 16; ++j) {              // D=64 -> 16 float4 pairs
        float4 a = ps[j], b = pd[j];
        acc += a.x * b.x + a.y * b.y + a.z * b.z + a.w * b.w;
    }
    acc *= 0.125f;                              // 1/sqrt(64)
    sc[idx] = acc;
    atomicMaxF(&mmax[d * heads + h], acc);
}

// ee = exp(e - m[dst]); atomic segment-sum into den
__global__ void edge_expnorm(const int* __restrict__ dst, float* __restrict__ sc,
                             const float* __restrict__ mmax, float* __restrict__ den,
                             int E, int heads) {
    int idx = blockIdx.x * blockDim.x + threadIdx.x;
    if (idx >= E * heads) return;
    int e = idx / heads, h = idx % heads;
    int d = dst[e];
    float ee = __expf(sc[idx] - mmax[d * heads + h]);
    sc[idx] = ee;
    atomicAdd(&den[d * heads + h], ee);
}

// One wave32 per (edge, head): a computed once (broadcast loads), lanes do
// coalesced row loads + 2 global f32 atomic adds each (D=64).
__global__ void edge_aggregate_wave(const float* __restrict__ ft, const int* __restrict__ src,
                                    const int* __restrict__ dst, const float* __restrict__ sc,
                                    const float* __restrict__ den, float* __restrict__ out,
                                    int E, int heads) {
    int wave = blockIdx.x * (blockDim.x >> 5) + (threadIdx.x >> 5);
    if (wave >= E * heads) return;
    int lane = threadIdx.x & 31;
    int e = wave / heads, h = wave % heads;
    int s = src[e], d = dst[e];
    float a = sc[wave] / den[d * heads + h];
    const float* __restrict__ ps = ft  + ((size_t)s * heads + h) * HID;
    float* __restrict__ po       = out + ((size_t)d * heads + h) * HID;
    atomicAdd(&po[lane],      ps[lane]      * a);
    atomicAdd(&po[lane + 32], ps[lane + 32] * a);
}

// ---------------- pooling + classifier ----------------
__global__ void pool_kernel(const float* __restrict__ x, const int* __restrict__ gid,
                            float* __restrict__ psum, float* __restrict__ cnt,
                            int n, int D) {
    int idx = blockIdx.x * blockDim.x + threadIdx.x;
    if (idx >= n * D) return;
    int node = idx / D, j = idx % D;
    int g = gid[node];
    atomicAdd(&psum[g * D + j], fmaxf(x[idx], 0.0f));   // final relu fused here
    if (j == 0) atomicAdd(&cnt[g], 1.0f);
}

__global__ void classifier_kernel(const float* __restrict__ psum, const float* __restrict__ cnt,
                                  const float* __restrict__ Wc, const float* __restrict__ bc,
                                  float* __restrict__ out) {
    int idx = blockIdx.x * blockDim.x + threadIdx.x;
    if (idx >= NUM_GRAPHS * N_CLASSES) return;
    int g = idx / N_CLASSES, c = idx % N_CLASSES;
    float inv = 1.0f / fmaxf(cnt[g], 1.0f);
    float acc = bc[c];
#pragma unroll 8
    for (int j = 0; j < HID; ++j)
        acc += psum[g * HID + j] * inv * Wc[j * N_CLASSES + c];
    out[idx] = acc;
}

// ---------------- launch ----------------
static inline int ceil_div(long long a, int b) { return (int)((a + b - 1) / b); }

extern "C" void kernel_launch(void* const* d_in, const int* in_sizes, int n_in,
                              void* d_out, int out_size, void* d_ws, size_t ws_size,
                              hipStream_t stream) {
    const float* h   = (const float*)d_in[0];
    const int*   src = (const int*)  d_in[1];
    const int*   dst = (const int*)  d_in[2];
    const int*   gid = (const int*)  d_in[3];
    const float* W1  = (const float*)d_in[4];   // [128, 256]
    const float* W2  = (const float*)d_in[5];   // [256, 64]
    const float* Wc  = (const float*)d_in[6];   // [64, 10]
    const float* bc  = (const float*)d_in[7];   // [10]
    float* out = (float*)d_out;                 // [128, 10]

    // ---- workspace carve-up: float region, then bf16 region ----
    float* ws   = (float*)d_ws;
    float* ft1  = ws;                             // N*256
    float* agg1 = ft1  + (size_t)NNODES * 256;    // N*256
    float* ft2  = agg1 + (size_t)NNODES * 256;    // N*64
    float* agg2 = ft2  + (size_t)NNODES * 64;     // N*64
    float* sc1  = agg2 + (size_t)NNODES * 64;     // E*4
    float* m1   = sc1  + (size_t)NEDGES * 4;      // N*4
    float* den1 = m1   + (size_t)NNODES * 4;      // N*4
    float* sc2  = den1 + (size_t)NNODES * 4;      // E
    float* m2   = sc2  + (size_t)NEDGES;          // N
    float* den2 = m2   + (size_t)NNODES;          // N
    float* psum = den2 + (size_t)NNODES;          // 128*64
    float* cnt  = psum + NUM_GRAPHS * HID;        // 128
    __bf16* hbf    = (__bf16*)(cnt + NUM_GRAPHS);           // N*128
    __bf16* w1t    = hbf  + (size_t)NNODES * IN_DIM;        // 256*128
    __bf16* agg1bf = w1t  + (size_t)(HEADS * HID) * IN_DIM; // N*256
    __bf16* w2t    = agg1bf + (size_t)NNODES * 256;         // 64*256

    const int BT = 256;
    const float NEG_INF = -INFINITY;

    // ---- init accumulated scratch ----
    fill_kernel<<<ceil_div((size_t)NNODES*256, BT), BT, 0, stream>>>(agg1, 0.0f, NNODES*256);
    fill_kernel<<<ceil_div((size_t)NNODES*64,  BT), BT, 0, stream>>>(agg2, 0.0f, NNODES*64);
    fill_kernel<<<ceil_div(NNODES*4, BT), BT, 0, stream>>>(m1,  NEG_INF, NNODES*4);
    fill_kernel<<<ceil_div(NNODES*4, BT), BT, 0, stream>>>(den1, 0.0f,   NNODES*4);
    fill_kernel<<<ceil_div(NNODES,   BT), BT, 0, stream>>>(m2,  NEG_INF, NNODES);
    fill_kernel<<<ceil_div(NNODES,   BT), BT, 0, stream>>>(den2, 0.0f,   NNODES);
    fill_kernel<<<ceil_div(NUM_GRAPHS*HID, BT), BT, 0, stream>>>(psum, 0.0f, NUM_GRAPHS*HID);
    fill_kernel<<<ceil_div(NUM_GRAPHS,     BT), BT, 0, stream>>>(cnt,  0.0f, NUM_GRAPHS);

    // ---- operand prep: bf16 A, bf16 transposed weights ----
    f32_to_bf16<false><<<ceil_div((long long)NNODES*IN_DIM, BT), BT, 0, stream>>>(
        h, hbf, (long long)NNODES * IN_DIM);
    transpose_to_bf16<<<ceil_div(IN_DIM * HEADS * HID, BT), BT, 0, stream>>>(
        W1, w1t, IN_DIM, HEADS * HID);
    transpose_to_bf16<<<ceil_div(HEADS * HID * HID, BT), BT, 0, stream>>>(
        W2, w2t, HEADS * HID, HID);

    // ---- layer 1: ft1 = h @ W1 ----
    {
        int waves = (NNODES / 16) * ((HEADS * HID) / 64);    // 1250 * 4
        wmma_gemm_bf16<<<ceil_div(waves, 8), 8 * 32, 0, stream>>>(
            hbf, w1t, ft1, NNODES, IN_DIM, HEADS * HID);
    }
    // ---- layer 1 edge softmax + aggregate (4 heads, D=64) ----
    edge_scores<<<ceil_div((long long)NEDGES*HEADS, BT), BT, 0, stream>>>(
        ft1, src, dst, sc1, m1, NEDGES, HEADS, HID);
    edge_expnorm<<<ceil_div((long long)NEDGES*HEADS, BT), BT, 0, stream>>>(
        dst, sc1, m1, den1, NEDGES, HEADS);
    edge_aggregate_wave<<<ceil_div((long long)NEDGES*HEADS, 8), 8 * 32, 0, stream>>>(
        ft1, src, dst, sc1, den1, agg1, NEDGES, HEADS);

    // ---- layer 2: ft2 = relu(agg1) @ W2 (relu fused into bf16 convert) ----
    f32_to_bf16<true><<<ceil_div((long long)NNODES*256, BT), BT, 0, stream>>>(
        agg1, agg1bf, (long long)NNODES * 256);
    {
        int waves = (NNODES / 16) * (HID / 64);              // 1250 * 1
        wmma_gemm_bf16<<<ceil_div(waves, 8), 8 * 32, 0, stream>>>(
            agg1bf, w2t, ft2, NNODES, HEADS * HID, HID);
    }
    // ---- layer 2 edge softmax + aggregate (1 head, D=64) ----
    edge_scores<<<ceil_div((long long)NEDGES, BT), BT, 0, stream>>>(
        ft2, src, dst, sc2, m2, NEDGES, 1, HID);
    edge_expnorm<<<ceil_div((long long)NEDGES, BT), BT, 0, stream>>>(
        dst, sc2, m2, den2, NEDGES, 1);
    edge_aggregate_wave<<<ceil_div((long long)NEDGES, 8), 8 * 32, 0, stream>>>(
        ft2, src, dst, sc2, den2, agg2, NEDGES, 1);

    // ---- mean pool (relu fused) + classifier ----
    pool_kernel<<<ceil_div((size_t)NNODES*HID, BT), BT, 0, stream>>>(
        agg2, gid, psum, cnt, NNODES, HID);
    classifier_kernel<<<ceil_div(NUM_GRAPHS*N_CLASSES, BT), BT, 0, stream>>>(
        psum, cnt, Wc, bc, out);
}
